// StringGNNDEGModel_6923487282087
// MI455X (gfx1250) — matrix operand order repaired and checked by reading, not compiled, not run on hardware
//
#include <hip/hip_runtime.h>
#include <hip/hip_bf16.h>

// ---------------------------------------------------------------------------
// StringGNNDEGModel on MI455X (gfx1250, wave32, WMMA + async-to-LDS)
//   transpose: W1/W2 f32 [512 x N] -> bf16 [N x 512]  (async-to-LDS tile flip)
//   prep:      gather + attention + layernorm -> x [1024 x 512] bf16
//   gemm1:     h = gelu(x @ W1 + b1)  -> hb bf16      (WMMA, LDS A-strip)
//   gemm2:     out = h @ W2 + b2      -> d_out f32    (WMMA, LDS A-strip)
// GEMM blocking: block = 8 waves, tile 64m x 384n; wave = 4m x 3n 16x16 tiles.
// A strip (64 x 512 bf16) staged once per block via global_load_async_to_lds.
// ---------------------------------------------------------------------------

#define HIDDEN 256
#define TWOH   512
#define KNBR   20
#define BATCH  1024
#define NOUT   19920   // 3 * 6640
#define HH     512

typedef __bf16 bf16_t;
typedef __attribute__((ext_vector_type(16))) __bf16 v16bf;
typedef __attribute__((ext_vector_type(8)))  float  v8f;

union Frag16 { v16bf v; uint4 q[2]; };

// LDS byte offset of a shared-memory object (generic ptr low 32 bits)
__device__ __forceinline__ unsigned lds_off(const void* p) {
    return (unsigned)(unsigned long long)p;
}
// CDNA5 async global->LDS copies (ASYNCcnt); GVS addressing: saddr + voffset
__device__ __forceinline__ void async_b32(unsigned lds, unsigned voff, const void* base) {
    asm volatile("global_load_async_to_lds_b32 %0, %1, %2"
                 :: "v"(lds), "v"(voff), "s"(base) : "memory");
}
__device__ __forceinline__ void async_b128(unsigned lds, unsigned voff, const void* base) {
    asm volatile("global_load_async_to_lds_b128 %0, %1, %2"
                 :: "v"(lds), "v"(voff), "s"(base) : "memory");
}
__device__ __forceinline__ void wait_async0() {
    asm volatile("s_wait_asynccnt 0x0" ::: "memory");
}

// ---------------------------------------------------------------------------
// block reduction (256 threads = 8 wave32); returns sum to all threads
// ---------------------------------------------------------------------------
__device__ __forceinline__ float block_reduce_sum(float v, float* s_red, int t) {
    int lane = t & 31, wid = t >> 5;
#pragma unroll
    for (int off = 16; off > 0; off >>= 1) v += __shfl_down(v, off, 32);
    __syncthreads();                 // guard s_red reuse across calls
    if (lane == 0) s_red[wid] = v;
    __syncthreads();
    float r = s_red[0];
#pragma unroll
    for (int w = 1; w < 8; ++w) r += s_red[w];
    return r;
}

// ---------------------------------------------------------------------------
// prep: one block (256 threads) per batch row; neighbor rows gathered into
// LDS with async-to-LDS b32 loads.
// ---------------------------------------------------------------------------
__global__ void __launch_bounds__(256) prep_kernel(
    const int*   __restrict__ node_idx,
    const int*   __restrict__ topk,
    const float* __restrict__ node_emb,
    const float* __restrict__ fallback,
    const float* __restrict__ w_attn,
    const float* __restrict__ gamma,
    const float* __restrict__ beta,
    bf16_t*      __restrict__ xb)          // [BATCH x 512] bf16
{
    __shared__ float s_nbr[KNBR * HIDDEN];   // 20 KB
    __shared__ float s_red[8];
    __shared__ float s_scores[KNBR];
    __shared__ float s_attn[KNBR];
    __shared__ float s_p0;

    const int b = blockIdx.x;
    const int t = threadIdx.x;

    const int  idx   = node_idx[b];
    const bool valid = idx >= 0;
    const int  safe  = valid ? idx : 0;

    float pert;
    {
        float g  = node_emb[(long)safe * HIDDEN + t];
        float fb = fallback[t];
        pert = valid ? (g + 1e-3f * fb) : fb;
    }
    if (t == 0) s_p0 = pert;
    const float wa = w_attn[t];

    // async gather of the K neighbor rows into LDS
    unsigned vmask = 0u;
    {
        const unsigned sbase = lds_off(s_nbr);
#pragma unroll 1
        for (int k = 0; k < KNBR; ++k) {
            int nidx = topk[safe * KNBR + k];
            if (nidx >= 0) vmask |= (1u << k);
            int cl = nidx > 0 ? nidx : 0;
            async_b32(sbase + (unsigned)(k * HIDDEN + t) * 4u,
                      (unsigned)(cl * HIDDEN + t) * 4u, node_emb);
        }
    }
    wait_async0();
    __syncthreads();

    // attention scores (masked dot with w_attn)
#pragma unroll 1
    for (int k = 0; k < KNBR; ++k) {
        float e = (vmask >> k) & 1u ? s_nbr[k * HIDDEN + t] : 0.f;
        float tot = block_reduce_sum(e * wa, s_red, t);
        if (t == 0) s_scores[k] = tot;
    }
    __syncthreads();

    if (t == 0) {
        float mx = -3.0e38f;
        float p0 = s_p0;
        for (int k = 0; k < KNBR; ++k) {
            float s = ((vmask >> k) & 1u) ? (p0 * s_scores[k]) : -1.0e9f;
            s_scores[k] = s;
            mx = s > mx ? s : mx;
        }
        float den = 0.f;
        for (int k = 0; k < KNBR; ++k) {
            float e = __expf(s_scores[k] - mx);
            s_attn[k] = e; den += e;
        }
        float inv = vmask ? (1.f / den) : 0.f;
        for (int k = 0; k < KNBR; ++k) s_attn[k] *= inv;
    }
    __syncthreads();

    float ctx = 0.f;
#pragma unroll
    for (int k = 0; k < KNBR; ++k) {
        float e = (vmask >> k) & 1u ? s_nbr[k * HIDDEN + t] : 0.f;
        ctx += s_attn[k] * e;
    }

    // LayerNorm over 512 values [pert(256) | ctx(256)]
    float sum = block_reduce_sum(pert + ctx, s_red, t);
    float sq  = block_reduce_sum(pert * pert + ctx * ctx, s_red, t);
    float mean = sum * (1.f / 512.f);
    float var  = sq * (1.f / 512.f) - mean * mean;
    float rstd = rsqrtf(var + 1e-5f);

    float x0 = (pert - mean) * rstd * gamma[t]          + beta[t];
    float x1 = (ctx  - mean) * rstd * gamma[HIDDEN + t] + beta[HIDDEN + t];
    xb[(long)b * TWOH + t]          = (bf16_t)x0;
    xb[(long)b * TWOH + HIDDEN + t] = (bf16_t)x1;
}

// ---------------------------------------------------------------------------
// tiled transpose+convert: f32 [512 x N] -> bf16 [N x 512]
// 64k x 64n tile via async-to-LDS; coalesced reads, 32B transposed stores.
// ---------------------------------------------------------------------------
__global__ void __launch_bounds__(256) transpose_f32_to_bf16t(
    const float* __restrict__ in, bf16_t* __restrict__ out, int N)
{
    __shared__ float s[64][68];            // padded vs bank conflicts
    const int t  = threadIdx.x;
    const int k0 = blockIdx.x * 64;        // K = 512 -> gridDim.x = 8
    const int n0 = blockIdx.y * 64;
    const unsigned sbase = lds_off(&s[0][0]);

    {
        int nn  = t & 63;
        int n   = n0 + nn;
        int ncl = n < N ? n : N - 1;       // clamp (results discarded on store)
        int kk0 = t >> 6;                  // 0..3
#pragma unroll
        for (int r = 0; r < 16; ++r) {
            int kk = kk0 + r * 4;
            async_b32(sbase + (unsigned)(kk * 68 + nn) * 4u,
                      (unsigned)((k0 + kk) * N + ncl) * 4u, in);
        }
    }
    wait_async0();
    __syncthreads();

    {
        int nn = t >> 2;                   // 0..63
        int n  = n0 + nn;
        if (n < N) {
            int koff = (t & 3) * 16;
            Frag16 o;
#pragma unroll
            for (int e = 0; e < 16; ++e) o.v[e] = (bf16_t)s[koff + e][nn];
            uint4* dst = (uint4*)(out + (long)n * 512 + k0 + koff);
            dst[0] = o.q[0];
            dst[1] = o.q[1];
        }
    }
}

// ---------------------------------------------------------------------------
// WMMA bf16 GEMM, K = 512. Block: 8 waves, tile 64m x 384n. Wave: 4m x 3n.
// A strip staged in LDS (async b128), B streamed from global (prefetched).
// ---------------------------------------------------------------------------
#define AROW 520   // padded LDS row stride (bf16): 1040B -> 4-bank lane stride

template <bool GELU>
__global__ void __launch_bounds__(256) gemm_wmma(
    const bf16_t* __restrict__ A,    // [M x 512] bf16, M = 64*gridDim.x
    const bf16_t* __restrict__ Bt,   // [N x 512] bf16 (pre-transposed weights)
    const float*  __restrict__ bias, // [N]
    float*        __restrict__ Cf,   // !GELU
    bf16_t*       __restrict__ Cb,   // GELU
    int N)
{
    __shared__ bf16_t s_a[64 * AROW];      // ~65 KB
    const int t    = threadIdx.x;
    const int wid  = t >> 5;
    const int lane = t & 31;
    const int half = lane >> 4;
    const int l16  = lane & 15;
    const int m0   = blockIdx.x * 64;
    const int nb0  = blockIdx.y * 384;

    // stage A strip [m0..m0+63] x [0..511] into LDS: 4096 x 16B chunks
    {
        const unsigned sbase = lds_off(s_a);
        const unsigned gbase = (unsigned)m0 * (512u * 2u);
#pragma unroll
        for (int it = 0; it < 16; ++it) {
            int c = it * 256 + t;
            int row = c >> 6, col = c & 63;
            async_b128(sbase + (unsigned)(row * (AROW * 2) + col * 16),
                       gbase + (unsigned)(row * 1024 + col * 16), A);
        }
    }
    wait_async0();
    __syncthreads();

    // wave tile columns (clamped; N is a multiple of 16 so validity is
    // uniform per 16-wide tile column -> EXEC stays all-ones for WMMA)
    int ncol[3];
    const bf16_t* Brow[3];
#pragma unroll
    for (int j = 0; j < 3; ++j) {
        ncol[j] = nb0 + wid * 48 + j * 16 + l16;
        int ncl = ncol[j] < N ? ncol[j] : N - 1;
        Brow[j] = Bt + (long)ncl * 512;
    }

    const v8f vzero = {0.f, 0.f, 0.f, 0.f, 0.f, 0.f, 0.f, 0.f};
    v8f acc[4][3];
#pragma unroll
    for (int i = 0; i < 4; ++i)
#pragma unroll
        for (int j = 0; j < 3; ++j) acc[i][j] = vzero;

    for (int kb = 0; kb < 512; kb += 32) {
        Frag16 af[4], bfr[3];
#pragma unroll
        for (int i = 0; i < 4; ++i) {      // A from LDS (ds_load_b128 x2)
            const uint4* A4 = (const uint4*)(&s_a[(i * 16 + l16) * AROW + kb]);
            af[i].q[0] = A4[half];         // K 0..7 / 8..15
            af[i].q[1] = A4[2 + half];     // K 16..23 / 24..31
        }
#pragma unroll
        for (int j = 0; j < 3; ++j) {      // B from global (b128 x2)
            const uint4* B4 = (const uint4*)(Brow[j] + kb);
            bfr[j].q[0] = B4[half];
            bfr[j].q[1] = B4[2 + half];
            if (kb + 32 < 512) __builtin_prefetch(Brow[j] + kb + 32, 0, 3);
        }
#pragma unroll
        for (int i = 0; i < 4; ++i)
#pragma unroll
            for (int j = 0; j < 3; ++j)
                acc[i][j] = __builtin_amdgcn_wmma_f32_16x16x32_bf16(
                    false, af[i].v, false, bfr[j].v,
                    (short)0, acc[i][j], false, false);
    }

    // C/D layout: VGPR r -> m = r + 8*half, n = l16
#pragma unroll
    for (int j = 0; j < 3; ++j) {
        if (ncol[j] >= N) continue;        // uniform per tile column
        float bv = bias[ncol[j]];
#pragma unroll
        for (int i = 0; i < 4; ++i) {
#pragma unroll
            for (int r = 0; r < 8; ++r) {
                int m = m0 + i * 16 + r + half * 8;
                float val = acc[i][j][r] + bv;
                if (GELU) {
                    val = 0.5f * val * (1.f + erff(val * 0.70710678118654752f));
                    Cb[(long)m * N + ncol[j]] = (bf16_t)val;
                } else {
                    Cf[(long)m * N + ncol[j]] = val;
                }
            }
        }
    }
}

// ---------------------------------------------------------------------------
extern "C" void kernel_launch(void* const* d_in, const int* in_sizes, int n_in,
                              void* d_out, int out_size, void* d_ws, size_t ws_size,
                              hipStream_t stream)
{
    const int*   node_idx = (const int*)  d_in[0];
    const int*   topk     = (const int*)  d_in[1];
    const float* node_emb = (const float*)d_in[2];
    const float* fallback = (const float*)d_in[3];
    const float* w_attn   = (const float*)d_in[4];
    const float* gamma    = (const float*)d_in[5];
    const float* beta     = (const float*)d_in[6];
    const float* W1       = (const float*)d_in[7];
    const float* b1       = (const float*)d_in[8];
    const float* W2       = (const float*)d_in[9];
    const float* b2       = (const float*)d_in[10];

    char* ws = (char*)d_ws;
    bf16_t* xb  = (bf16_t*)(ws);                              // 1 MB  [1024x512]
    bf16_t* hb  = (bf16_t*)(ws + (1u << 20));                 // 1 MB  [1024x512]
    bf16_t* W1t = (bf16_t*)(ws + (2u << 20));                 // 0.5MB [512x512]
    bf16_t* W2t = (bf16_t*)(ws + (2u << 20) + 524288u);       // 20 MB [19920x512]

    // weight convert+transpose (per-call; deterministic, capture-safe)
    transpose_f32_to_bf16t<<<dim3(8, (512  + 63) / 64), 256, 0, stream>>>(W1, W1t, 512);
    transpose_f32_to_bf16t<<<dim3(8, (NOUT + 63) / 64), 256, 0, stream>>>(W2, W2t, NOUT);

    prep_kernel<<<BATCH, 256, 0, stream>>>(node_idx, topk, node_emb, fallback,
                                           w_attn, gamma, beta, xb);

    // GEMM1 + GELU: [1024x512]*[512x512] -> hb bf16
    gemm_wmma<true><<<dim3(BATCH / 64, (HH + 383) / 384), 256, 0, stream>>>(
        xb, W1t, b1, nullptr, hb, HH);

    // GEMM2: [1024x512]*[512x19920] -> d_out f32
    gemm_wmma<false><<<dim3(BATCH / 64, (NOUT + 383) / 384), 256, 0, stream>>>(
        hb, W2t, b2, (float*)d_out, nullptr, NOUT);
}
